// ShiftedWindowAttention_74887049773493
// MI455X (gfx1250) — matrix operand order repaired and checked by reading, not compiled, not run on hardware
//
#include <hip/hip_runtime.h>
#include <hip/hip_bf16.h>

typedef __attribute__((ext_vector_type(16))) _Float16 v16h;
typedef __attribute__((ext_vector_type(8)))  _Float16 v8h;
typedef __attribute__((ext_vector_type(8)))  float    v8f;
typedef __attribute__((ext_vector_type(4)))  float    v4f;

#define HEADS    12
#define HEAD_DIM 32
#define CDIM     384
#define QKV_N    1152
#define MTOT     131072   // B * H * W = 32 * 64 * 64

__device__ __forceinline__ v8f vzero8() {
  v8f z;
#pragma unroll
  for (int i = 0; i < 8; ++i) z[i] = 0.0f;
  return z;
}

__device__ __forceinline__ v16h vcat(v8h lo, v8h hi) {
  return __builtin_shufflevector(lo, hi, 0, 1, 2, 3, 4, 5, 6, 7,
                                 8, 9, 10, 11, 12, 13, 14, 15);
}

__device__ __forceinline__ v8f wmma16x16x32(v16h a, v16h b, v8f c) {
  return __builtin_amdgcn_wmma_f32_16x16x32_f16(false, a, false, b,
                                                (short)0, c, false, false);
}

// Async global->LDS 16-byte copy (no VGPR data path, tracked by ASYNCcnt).
// lds_off: byte offset within the workgroup's LDS allocation (low 32 bits of
// the generic pointer); gaddr: 64-bit global byte address.
__device__ __forceinline__ void async_copy16(uint32_t lds_off, uint64_t gaddr) {
  asm volatile("global_load_async_to_lds_b128 %0, %1, off"
               :: "v"(lds_off), "v"(gaddr) : "memory");
}

__device__ __forceinline__ void wait_async0() {
  asm volatile("s_wait_asynccnt 0x0" ::: "memory");
}

// A fragment (16x32 f16) from a row-major LDS tile.
// Lane l<16: row m0+l, K-halves [k0..k0+7] and [k0+16..k0+23];
// lane l>=16: same rows, K-halves [k0+8..] and [k0+24..].
__device__ __forceinline__ v16h load_a_frag(const _Float16* base, int m0,
                                            int k0, int ld, int lane) {
  const _Float16* p = base + (m0 + (lane & 15)) * ld + k0 + ((lane >> 4) << 3);
  v8h lo = *(const v8h*)(p);
  v8h hi = *(const v8h*)(p + 16);
  return vcat(lo, hi);
}

// B fragment (32x16 f16): LDS holds B^T row-major (one B-column per row).
// Lane l<16: col n0+l, K-halves [k0..k0+7] and [k0+16..k0+23];
// lane l>=16: same cols, K-halves [k0+8..] and [k0+24..].
__device__ __forceinline__ v16h load_b_frag(const _Float16* base, int n0,
                                            int k0, int ld, int lane) {
  const _Float16* p = base + (n0 + (lane & 15)) * ld + k0 + ((lane >> 4) << 3);
  v8h lo = *(const v8h*)(p);
  v8h hi = *(const v8h*)(p + 16);
  return vcat(lo, hi);
}

// ---------------------------------------------------------------------------
// Kernel 1: qkv = f16( (x @ qkv_w + qkv_b) * (n<384 ? 1/sqrt(32) : 1) )
// M=131072, K=384, N=1152. Block tile 128x128, K-step 32, 8 waves (4Mx2N),
// each wave 32x64 -> 2x4 accumulators of 16x16.
// ---------------------------------------------------------------------------
__global__ __launch_bounds__(256) void qkv_gemm_kernel(
    const float* __restrict__ X, const float* __restrict__ W,
    const float* __restrict__ bias, _Float16* __restrict__ out) {
  __shared__ alignas(16) _Float16 Ash[128 * 40];
  __shared__ alignas(16) _Float16 Bsh[128 * 40];
  const int t = threadIdx.x, lane = t & 31, wid = t >> 5;
  const int mBase = blockIdx.x * 128;
  const int nBase = blockIdx.y * 128;
  const int wm = (wid & 3) * 32;
  const int wn = (wid >> 2) * 64;

  v8f acc[2][4];
#pragma unroll
  for (int i = 0; i < 2; ++i)
#pragma unroll
    for (int j = 0; j < 4; ++j) acc[i][j] = vzero8();

  const int arow = t >> 1, aseg = (t & 1) << 4;

  for (int kb = 0; kb < CDIM; kb += 32) {
    {  // stage A: 128x32 fp32 -> f16 (row-major)
      const float* src = X + (size_t)(mBase + arow) * CDIM + kb + aseg;
      v4f f0 = *(const v4f*)(src);
      v4f f1 = *(const v4f*)(src + 4);
      v4f f2 = *(const v4f*)(src + 8);
      v4f f3 = *(const v4f*)(src + 12);
      v8h h0, h1;
#pragma unroll
      for (int i = 0; i < 4; ++i) {
        h0[i]     = (_Float16)f0[i];
        h0[i + 4] = (_Float16)f1[i];
        h1[i]     = (_Float16)f2[i];
        h1[i + 4] = (_Float16)f3[i];
      }
      *(v8h*)(&Ash[arow * 40 + aseg])     = h0;
      *(v8h*)(&Ash[arow * 40 + aseg + 8]) = h1;
      if (kb + 32 < CDIM) __builtin_prefetch(src + 32, 0, 1);
    }
    {  // stage B transposed: Bsh[n][k] = W[kb+k][nBase+n], fp32 -> f16
      const float* src = W + (size_t)(kb + aseg) * QKV_N + nBase + arow;
      v8h h0, h1;
#pragma unroll
      for (int j = 0; j < 8; ++j) {
        h0[j] = (_Float16)src[(size_t)j * QKV_N];
        h1[j] = (_Float16)src[(size_t)(j + 8) * QKV_N];
      }
      *(v8h*)(&Bsh[arow * 40 + aseg])     = h0;
      *(v8h*)(&Bsh[arow * 40 + aseg + 8]) = h1;
      if (kb + 32 < CDIM) __builtin_prefetch(src + (size_t)32 * QKV_N, 0, 1);
    }
    __syncthreads();
    v16h a0 = load_a_frag(Ash, wm, 0, 40, lane);
    v16h a1 = load_a_frag(Ash, wm + 16, 0, 40, lane);
#pragma unroll
    for (int nt = 0; nt < 4; ++nt) {
      v16h b = load_b_frag(Bsh, wn + nt * 16, 0, 40, lane);
      acc[0][nt] = wmma16x16x32(a0, b, acc[0][nt]);
      acc[1][nt] = wmma16x16x32(a1, b, acc[1][nt]);
    }
    __syncthreads();
  }

  const int ccol = lane & 15, roff = (lane >> 4) << 3;
#pragma unroll
  for (int mt = 0; mt < 2; ++mt)
#pragma unroll
    for (int nt = 0; nt < 4; ++nt) {
      int nG = nBase + wn + nt * 16 + ccol;
      float bv = bias[nG];
      float scl = (nG < CDIM) ? 0.17677669529663687f : 1.0f;  // q * d^-0.5
#pragma unroll
      for (int r = 0; r < 8; ++r) {
        int mG = mBase + wm + mt * 16 + r + roff;
        out[(size_t)mG * QKV_N + nG] = (_Float16)((acc[mt][nt][r] + bv) * scl);
      }
    }
}

// ---------------------------------------------------------------------------
// Kernel 2: expand relative-position bias to biasH[head][m][n] (12x64x64 f32)
// ---------------------------------------------------------------------------
__global__ void bias_expand_kernel(const float* __restrict__ table,
                                   const int* __restrict__ ridx,
                                   float* __restrict__ biasH) {
  int idx = blockIdx.x * 256 + threadIdx.x;
  if (idx < HEADS * 4096) {
    int h = idx >> 12, mn = idx & 4095;
    biasH[idx] = table[(size_t)ridx[mn] * HEADS + h];
  }
}

// ---------------------------------------------------------------------------
// Kernel 3: shifted-window attention, one workgroup per (window, head).
// 4 waves; wave w computes attn rows 16w..16w+15. Token gather applies the
// cyclic shift; mask regions computed analytically; scatter applies inverse.
// Q/K tiles are staged with async global->LDS copies (ASYNCcnt); V is
// transposed through VGPRs into LDS.
// ---------------------------------------------------------------------------
__global__ __launch_bounds__(128) void swin_attn_kernel(
    const _Float16* __restrict__ qkv, const float* __restrict__ biasH,
    _Float16* __restrict__ outh) {
  __shared__ alignas(16) _Float16 Qs[64 * 40];  // row-major (tok, d)
  __shared__ alignas(16) _Float16 Ks[64 * 40];  // row-major (tok, d)
  __shared__ alignas(16) _Float16 Vt[32 * 72];  // transposed (d, tok)
  __shared__ alignas(16) _Float16 Ps[64 * 72];  // softmax(P) f16 (tok, tok)
  const int t = threadIdx.x, lane = t & 31, wid = t >> 5;
  const int head = blockIdx.x % HEADS;
  const int win = blockIdx.x / HEADS;
  const int b = win >> 6, w = win & 63, wy = w >> 3, wx = w & 7;

  // stage Q, K (async), V transposed; 64 rows x 4 chunks of 8 halves
  for (int c = t; c < 256; c += 128) {
    int m = c >> 2, s = (c & 3) << 3;
    int sy = (wy << 3) + (m >> 3), sx = (wx << 3) + (m & 7);
    int tokRow = (b << 12) + (((sy + 4) & 63) << 6) + ((sx + 4) & 63);
    const _Float16* src = qkv + (size_t)tokRow * QKV_N + head * HEAD_DIM + s;
    async_copy16((uint32_t)(uintptr_t)&Qs[m * 40 + s],
                 (uint64_t)(uintptr_t)src);
    async_copy16((uint32_t)(uintptr_t)&Ks[m * 40 + s],
                 (uint64_t)(uintptr_t)(src + CDIM));
    v8h vv = *(const v8h*)(src + 2 * CDIM);
#pragma unroll
    for (int j = 0; j < 8; ++j) Vt[(s + j) * 72 + m] = vv[j];
  }
  wait_async0();
  __syncthreads();

  const int m0 = wid << 4;
  // S = Q K^T  (q pre-scaled); B-matrix of QK^T is K^T -> K rows contiguous
  v16h aq = load_a_frag(Qs, m0, 0, 40, lane);
  v8f att[4];
#pragma unroll
  for (int j = 0; j < 4; ++j) {
    v16h bk = load_b_frag(Ks, j << 4, 0, 40, lane);
    att[j] = wmma16x16x32(aq, bk, vzero8());
  }

  const int ccol = lane & 15, roff = (lane >> 4) << 3;
  const bool edge = (wy == 7) || (wx == 7);
  int regcol[4];
  if (edge) {
#pragma unroll
    for (int j = 0; j < 4; ++j) {
      int cm = (j << 4) + ccol;
      int sy = (wy << 3) + (cm >> 3), sx = (wx << 3) + (cm & 7);
      regcol[j] = (sy < 56 ? 0 : (sy < 60 ? 1 : 2)) * 3 +
                  (sx < 56 ? 0 : (sx < 60 ? 1 : 2));
    }
  }

#pragma unroll
  for (int r = 0; r < 8; ++r) {
    int m = m0 + r + roff;
    int regm = 0;
    if (edge) {
      int sy = (wy << 3) + (m >> 3), sx = (wx << 3) + (m & 7);
      regm = (sy < 56 ? 0 : (sy < 60 ? 1 : 2)) * 3 +
             (sx < 56 ? 0 : (sx < 60 ? 1 : 2));
    }
    float mx = -1e30f;
#pragma unroll
    for (int j = 0; j < 4; ++j) {
      float v = att[j][r] + biasH[head * 4096 + m * 64 + (j << 4) + ccol];
      if (edge && regcol[j] != regm) v -= 100.0f;
      att[j][r] = v;
      mx = fmaxf(mx, v);
    }
#pragma unroll
    for (int d = 8; d >= 1; d >>= 1) mx = fmaxf(mx, __shfl_xor(mx, d, 32));
    float sum = 0.0f;
#pragma unroll
    for (int j = 0; j < 4; ++j) {
      float e = __expf(att[j][r] - mx);
      att[j][r] = e;
      sum += e;
    }
#pragma unroll
    for (int d = 8; d >= 1; d >>= 1) sum += __shfl_xor(sum, d, 32);
    float inv = 1.0f / sum;
#pragma unroll
    for (int j = 0; j < 4; ++j)
      Ps[m * 72 + (j << 4) + ccol] = (_Float16)(att[j][r] * inv);
  }
  // same-wave LDS RAW: DS ops are in-order; wait before fragment reload
  asm volatile("s_wait_dscnt 0x0" ::: "memory");

  // O = P V : K=64 in two 32-chunks; V transposed so B-frag is contiguous
#pragma unroll
  for (int nt = 0; nt < 2; ++nt) {
    v8f o = vzero8();
#pragma unroll
    for (int kc = 0; kc < 2; ++kc) {
      v16h ap = load_a_frag(Ps, m0, kc << 5, 72, lane);
      v16h bv = load_b_frag(Vt, nt << 4, kc << 5, 72, lane);
      o = wmma16x16x32(ap, bv, o);
    }
#pragma unroll
    for (int r = 0; r < 8; ++r) {
      int m = m0 + r + roff;
      int sy = (wy << 3) + (m >> 3), sx = (wx << 3) + (m & 7);
      int tokRow = (b << 12) + (((sy + 4) & 63) << 6) + ((sx + 4) & 63);
      outh[(size_t)tokRow * CDIM + head * HEAD_DIM + (nt << 4) + ccol] =
          (_Float16)o[r];
    }
  }
}

// ---------------------------------------------------------------------------
// Kernel 4: out = attnout(f16) @ proj_w + proj_b, fp32 output to d_out.
// M=131072, K=384, N=384. A-tile staged via async global->LDS copies.
// ---------------------------------------------------------------------------
__global__ __launch_bounds__(256) void proj_gemm_kernel(
    const _Float16* __restrict__ A, const float* __restrict__ W,
    const float* __restrict__ bias, float* __restrict__ out) {
  __shared__ alignas(16) _Float16 Ash[128 * 40];
  __shared__ alignas(16) _Float16 Bsh[128 * 40];
  const int t = threadIdx.x, lane = t & 31, wid = t >> 5;
  const int mBase = blockIdx.x * 128;
  const int nBase = blockIdx.y * 128;
  const int wm = (wid & 3) * 32;
  const int wn = (wid >> 2) * 64;

  v8f acc[2][4];
#pragma unroll
  for (int i = 0; i < 2; ++i)
#pragma unroll
    for (int j = 0; j < 4; ++j) acc[i][j] = vzero8();

  const int arow = t >> 1, aseg = (t & 1) << 4;

  for (int kb = 0; kb < CDIM; kb += 32) {
    {  // stage A (already f16): async direct-to-LDS, no VGPR round trip
      const _Float16* src = A + (size_t)(mBase + arow) * CDIM + kb + aseg;
      uint32_t dst = (uint32_t)(uintptr_t)&Ash[arow * 40 + aseg];
      async_copy16(dst, (uint64_t)(uintptr_t)src);
      async_copy16(dst + 16, (uint64_t)(uintptr_t)(src + 8));
      if (kb + 32 < CDIM) __builtin_prefetch(src + 32, 0, 1);
    }
    {  // stage B transposed fp32 -> f16
      const float* src = W + (size_t)(kb + aseg) * CDIM + nBase + arow;
      v8h h0, h1;
#pragma unroll
      for (int j = 0; j < 8; ++j) {
        h0[j] = (_Float16)src[(size_t)j * CDIM];
        h1[j] = (_Float16)src[(size_t)(j + 8) * CDIM];
      }
      *(v8h*)(&Bsh[arow * 40 + aseg])     = h0;
      *(v8h*)(&Bsh[arow * 40 + aseg + 8]) = h1;
      if (kb + 32 < CDIM) __builtin_prefetch(src + (size_t)32 * CDIM, 0, 1);
    }
    wait_async0();
    __syncthreads();
    v16h a0 = load_a_frag(Ash, wm, 0, 40, lane);
    v16h a1 = load_a_frag(Ash, wm + 16, 0, 40, lane);
#pragma unroll
    for (int nt = 0; nt < 4; ++nt) {
      v16h b = load_b_frag(Bsh, wn + nt * 16, 0, 40, lane);
      acc[0][nt] = wmma16x16x32(a0, b, acc[0][nt]);
      acc[1][nt] = wmma16x16x32(a1, b, acc[1][nt]);
    }
    __syncthreads();
  }

  const int ccol = lane & 15, roff = (lane >> 4) << 3;
#pragma unroll
  for (int mt = 0; mt < 2; ++mt)
#pragma unroll
    for (int nt = 0; nt < 4; ++nt) {
      int nG = nBase + wn + nt * 16 + ccol;
      float bv = bias[nG];
#pragma unroll
      for (int r = 0; r < 8; ++r) {
        int mG = mBase + wm + mt * 16 + r + roff;
        out[(size_t)mG * CDIM + nG] = acc[mt][nt][r] + bv;
      }
    }
}

// ---------------------------------------------------------------------------
extern "C" void kernel_launch(void* const* d_in, const int* in_sizes, int n_in,
                              void* d_out, int out_size, void* d_ws,
                              size_t ws_size, hipStream_t stream) {
  const float* x          = (const float*)d_in[0];
  const float* qkv_w      = (const float*)d_in[1];
  const float* qkv_b      = (const float*)d_in[2];
  const float* proj_w     = (const float*)d_in[3];
  const float* proj_b     = (const float*)d_in[4];
  const float* bias_table = (const float*)d_in[5];
  const int*   rel_index  = (const int*)d_in[6];
  float* out = (float*)d_out;

  const size_t qkvBytes = (size_t)MTOT * QKV_N * sizeof(_Float16);  // ~302 MB
  const size_t attBytes = (size_t)MTOT * CDIM * sizeof(_Float16);   // ~101 MB
  _Float16* qkvh    = (_Float16*)d_ws;
  _Float16* attnout = (_Float16*)((char*)d_ws + qkvBytes);
  float*    biasH   = (float*)((char*)d_ws + qkvBytes + attBytes);

  hipLaunchKernelGGL(bias_expand_kernel, dim3(192), dim3(256), 0, stream,
                     bias_table, rel_index, biasH);
  hipLaunchKernelGGL(qkv_gemm_kernel, dim3(1024, 9), dim3(256), 0, stream,
                     x, qkv_w, qkv_b, qkvh);
  hipLaunchKernelGGL(swin_attn_kernel, dim3(2048 * HEADS), dim3(128), 0,
                     stream, qkvh, biasH, attnout);
  hipLaunchKernelGGL(proj_gemm_kernel, dim3(1024, 3), dim3(256), 0, stream,
                     attnout, proj_w, proj_b, out);
}
